// Projected_Adaptor_37151467111011
// MI455X (gfx1250) — matrix operand order repaired and checked by reading, not compiled, not run on hardware
//
#include <hip/hip_runtime.h>

typedef __attribute__((ext_vector_type(16))) __bf16 v16bf;
typedef __attribute__((ext_vector_type(8)))  __bf16 v8bf;
typedef __attribute__((ext_vector_type(8)))  float  v8f;
typedef __attribute__((ext_vector_type(4)))  float  v4f;

#define E_DIM 768
#define T_DIM 128
#define B_DIM 2
#define V_DIM 50257
#define M_DIM (B_DIM * T_DIM)
#define EPS 0.01f

// ---------------------------------------------------------------------------
// Kernel 1: adapted_state[b,t,f] = state + eps * sum_{s,r} sw[b,s]*u[b,t,s,r]*p1[s,f,r]
// where u[b,t,s,r] = sum_e state[b,t,e]*p2[s,e,r].  Output in bf16 (WMMA A operand).
// One block per (b,t) row; trivial FLOPs (~13 MFLOP total).
// ---------------------------------------------------------------------------
__global__ __launch_bounds__(256) void prep_adapted(
    const float* __restrict__ state, const float* __restrict__ sw,
    const float* __restrict__ p1, const float* __restrict__ p2,
    __bf16* __restrict__ As)
{
    __shared__ float srow[E_DIM];
    __shared__ float partial[16][17];
    __shared__ float coef[16];
    const int row = blockIdx.x;            // b*T + t
    const int b   = row / T_DIM;
    const int tid = threadIdx.x;

    for (int i = tid; i < E_DIM; i += 256)
        srow[i] = state[(size_t)row * E_DIM + i];
    __syncthreads();

    // 16 groups of 16 threads; group g handles (s,r) = (g>>3, g&7)
    const int g   = tid >> 4;
    const int l16 = tid & 15;
    const int s   = g >> 3;
    const int r   = g & 7;
    float acc = 0.f;
    for (int e = l16; e < E_DIM; e += 16)
        acc += srow[e] * p2[((size_t)s * E_DIM + e) * 8 + r];
    partial[g][l16] = acc;
    __syncthreads();

    if (tid < 16) {
        float sum = 0.f;
        #pragma unroll
        for (int i = 0; i < 16; ++i) sum += partial[tid][i];
        coef[tid] = EPS * sw[b * 2 + (tid >> 3)] * sum;
    }
    __syncthreads();

    for (int f = tid; f < E_DIM; f += 256) {
        float v = srow[f];
        #pragma unroll
        for (int gg = 0; gg < 16; ++gg)
            v += coef[gg] * p1[(((size_t)(gg >> 3)) * E_DIM + f) * 8 + (gg & 7)];
        As[(size_t)row * E_DIM + f] = (__bf16)v;
    }
}

// ---------------------------------------------------------------------------
// Kernel 2: logits[m, n] = sum_k adapted[m,k] * W[n,k]
// M=256, N=50257, K=768.
// Block = 4 waves sharing ONE 16-wide N tile; wave w covers M rows
// [w*64, w*64+64) as 4 accumulator tiles.  All 4 waves issue identical W
// loads at the same time -> one HBM fetch per W line (single-pass W, 154 MB),
// the rest hit WGP$/L2.  B fragment is register-reused across 4 WMMAs.
// ---------------------------------------------------------------------------
__global__ __launch_bounds__(128) void gemm_wmma(
    const __bf16* __restrict__ As, const float* __restrict__ W,
    float* __restrict__ out)
{
    const int lane  = threadIdx.x & 31;
    const int wave  = threadIdx.x >> 5;
    const int n0    = blockIdx.x * 16;     // N tile (shared by all 4 waves)
    const int mbase = wave * 64;           // M strip: 4 tiles of 16 rows
    const int l15   = lane & 15;
    const int hi    = lane >> 4;

    // B operand: lane n holds 16 consecutive K at offset 16*hi (clamp OOB rows)
    int nrow = n0 + l15;
    if (nrow > V_DIM - 1) nrow = V_DIM - 1;
    const int kbB = hi * 16;
    // A operand: lane m holds 16 consecutive K at offset 16*hi
    const int kbA = hi * 16;

    v8f zero = {0.f, 0.f, 0.f, 0.f, 0.f, 0.f, 0.f, 0.f};
    v8f c[4];
    #pragma unroll
    for (int j = 0; j < 4; ++j) c[j] = zero;

    const float*  wbase = W  + (size_t)nrow * E_DIM + kbB;
    const __bf16* abase = As + (size_t)(mbase + l15) * E_DIM + kbA;

    for (int k0 = 0; k0 < E_DIM; k0 += 32) {
        const float* wp = wbase + k0;
        // Regular-temporal loads: 3 of 4 waves hit near cache on these lines.
        v4f b0 = *(const v4f*)(wp + 0);
        v4f b1 = *(const v4f*)(wp + 4);
        v4f b2 = *(const v4f*)(wp + 8);
        v4f b3 = *(const v4f*)(wp + 12);
        __builtin_prefetch((const void*)(wp + 32), 0, 1);   // next K-step of W

        v16bf bf;
        #pragma unroll
        for (int i = 0; i < 4; ++i) {
            bf[i]      = (__bf16)b0[i];
            bf[4 + i]  = (__bf16)b1[i];
            bf[8 + i]  = (__bf16)b2[i];
            bf[12 + i] = (__bf16)b3[i];
        }

        #pragma unroll
        for (int j = 0; j < 4; ++j) {
            const __bf16* ap = abase + (size_t)j * 16 * E_DIM + k0;
            v8bf alo = *(const v8bf*)(ap);
            v8bf ahi = *(const v8bf*)(ap + 8);
            v16bf af = __builtin_shufflevector(alo, ahi,
                0, 1, 2, 3, 4, 5, 6, 7, 8, 9, 10, 11, 12, 13, 14, 15);
            c[j] = __builtin_amdgcn_wmma_f32_16x16x32_bf16(
                false, af, false, bf, (short)0, c[j], false, false);
        }
    }

    // D layout: VGPR q -> row 8*hi + q within tile, col = l15
    const int colg = n0 + l15;
    if (colg < V_DIM) {
        const int rbase = mbase + hi * 8;
        #pragma unroll
        for (int j = 0; j < 4; ++j) {
            #pragma unroll
            for (int q = 0; q < 8; ++q) {
                __builtin_nontemporal_store(
                    c[j][q],
                    out + (size_t)(rbase + j * 16 + q) * V_DIM + colg);
            }
        }
    }
}

// ---------------------------------------------------------------------------
extern "C" void kernel_launch(void* const* d_in, const int* in_sizes, int n_in,
                              void* d_out, int out_size, void* d_ws, size_t ws_size,
                              hipStream_t stream) {
    const float* state = (const float*)d_in[0];   // [B,T,E]
    const float* sw    = (const float*)d_in[1];   // [B,S]
    const float* W     = (const float*)d_in[2];   // [V,E]
    const float* p1    = (const float*)d_in[3];   // [S,E,R]
    const float* p2    = (const float*)d_in[4];   // [S,E,R]
    float* out = (float*)d_out;                   // [B,T,V]
    __bf16* As = (__bf16*)d_ws;                   // [M, E] bf16 adapted state

    prep_adapted<<<dim3(M_DIM), dim3(256), 0, stream>>>(state, sw, p1, p2, As);

    const int ntiles = (V_DIM + 15) / 16;         // 3142 N tiles, 1 block each
    gemm_wmma<<<dim3(ntiles), dim3(128), 0, stream>>>(As, W, out);
}
